// Attention_15255723835474
// MI455X (gfx1250) — matrix operand order repaired and checked by reading, not compiled, not run on hardware
//
#include <hip/hip_runtime.h>
#include <hip/hip_bf16.h>

// ---------------------------------------------------------------------------
// Attention layer for MI455X (gfx1250, wave32, WMMA, TDM tile copies).
// Pipeline: f32->bf16 convert, QKV GEMMs (bf16 WMMA, f32 acc), RoPE,
// flash-attention (online softmax, WMMA for QK^T and PV), output GEMM (f32 out).
// ---------------------------------------------------------------------------

typedef __bf16 bf16;
typedef __attribute__((ext_vector_type(16))) __bf16 bf16x16;
typedef __attribute__((ext_vector_type(8)))  float  floatx8;
typedef __attribute__((ext_vector_type(4)))  unsigned int uint32x4;
typedef __attribute__((ext_vector_type(8)))  int int32x8;
typedef __attribute__((ext_vector_type(4)))  int int32x4;

#define NHEADS   16
#define KVHEADS  4
#define HEADDIM  128

#if __has_builtin(__builtin_amdgcn_tensor_load_to_lds)
#define HAVE_TDM 1
#else
#define HAVE_TDM 0
#endif

union Frag {
  bf16x16 v;
  uint4   q[2];
  bf16    e[16];
};

static __device__ __forceinline__ floatx8 wmma_bf16(bf16x16 a, bf16x16 b, floatx8 c) {
  return __builtin_amdgcn_wmma_f32_16x16x32_bf16(
      false, a, false, b, (short)0, c, false, false);
}

// ---- Fallback: per-lane async memory->LDS copy (ASYNCcnt) ------------------
static __device__ __forceinline__ void async_copy_b128(const bf16* lds_dst, const bf16* gsrc) {
  uint32_t l = (uint32_t)(uintptr_t)lds_dst;
  unsigned long long g = (unsigned long long)(uintptr_t)gsrc;
  asm volatile("global_load_async_to_lds_b128 %0, %1, off" :: "v"(l), "v"(g) : "memory");
}
static __device__ __forceinline__ void wait_async0() {
  asm volatile("s_wait_asynccnt 0x0" ::: "memory");
}

#if HAVE_TDM
// ---- Tensor Data Mover: one instruction moves a strided 2-D bf16 tile ------
// D# per CDNA5 ISA ch8: group0 = {count=1, lds_addr, global_addr, type=2};
// group1 = {mask=0, data_size=1(2B), dims, strides (element units)}.
// Rows are packed contiguously in LDS (row stride = tile_d0 elements).
static __device__ __forceinline__ void tdm_load_2d_bf16(
    const bf16* lds_dst, const bf16* gsrc,
    uint32_t tile_d0, uint32_t tile_d1, uint64_t stride0_elems)
{
  uint64_t ga  = (uint64_t)(uintptr_t)gsrc;
  uint32_t lds = (uint32_t)(uintptr_t)lds_dst;

  union { uint64_t u[2]; uint32x4 v; } G0;
  G0.u[0] = 1ull | ((uint64_t)lds << 32);                  // count=1 | lds_addr
  G0.u[1] = (ga & ((1ull << 57) - 1)) | (2ull << 62);      // global_addr | type=2

  union { uint64_t u[4]; int32x8 v; } G1;
  // w0: [17:16]=data_size(1=2B), [63:48]=tensor_dim0[15:0]
  G1.u[0] = (1ull << 16) | ((uint64_t)(tile_d0 & 0xffff) << 48);
  // w1: [15:0]=tensor_dim0[31:16], [47:16]=tensor_dim1, [63:48]=tile_dim0
  G1.u[1] = (uint64_t)(tile_d0 >> 16)
          | ((uint64_t)tile_d1 << 16)
          | ((uint64_t)(tile_d0 & 0xffff) << 48);
  // w2: [15:0]=tile_dim1, [31:16]=tile_dim2(0), [63:32]=stride0[31:0]
  G1.u[2] = (uint64_t)(tile_d1 & 0xffff)
          | ((stride0_elems & 0xffffffffull) << 32);
  // w3: [15:0]=stride0[47:32], rest (dim1 stride, unused for 2-D tile)=0
  G1.u[3] = (stride0_elems >> 32) & 0xffffull;

  int32x4 z4 = {};
#if defined(__clang_major__) && __clang_major__ >= 23
  int32x8 z8 = {};
  __builtin_amdgcn_tensor_load_to_lds(G0.v, G1.v, z4, z4, z8, 0);
#else
  __builtin_amdgcn_tensor_load_to_lds(G0.v, G1.v, z4, z4, 0);
#endif
}
#endif

static __device__ __forceinline__ void wait_tile_copies() {
#if HAVE_TDM
  __builtin_amdgcn_s_wait_tensorcnt((short)0);
#else
  wait_async0();
#endif
}

// ---------------------------------------------------------------------------
// f32 -> bf16 convert
// ---------------------------------------------------------------------------
__global__ __launch_bounds__(256) void f32_to_bf16_kernel(
    const float* __restrict__ in, bf16* __restrict__ out, size_t n)
{
  size_t i = (size_t)blockIdx.x * blockDim.x + threadIdx.x;
  if (i < n) out[i] = (bf16)in[i];
}

// ---------------------------------------------------------------------------
// GEMM: C[M,N] = A[M,K] * B[K,N], bf16 in, f32-acc WMMA, C f32 or bf16.
// Block tile 128x128, BK=32, double-buffered LDS. 8 waves: 4(M) x 2(N),
// each wave 32x64 = 2x4 WMMA tiles (8 wmma / k-step).
// A tile: TDM (wave 0 issues one tensor_load_to_lds per tile).
// B tile: VGPR loads issued early, scatter-transposed into LDS after WMMA.
// ---------------------------------------------------------------------------
template <bool OUT_F32>
__global__ __launch_bounds__(256) void gemm_bf16_kernel(
    const bf16* __restrict__ A, const bf16* __restrict__ Bm,
    void* __restrict__ C, int M, int N, int K)
{
  constexpr int BM = 128, BN = 128, BK = 32;
  __shared__ bf16 sA[2][BM * BK];     // [m][k]            2 x 8 KB
  __shared__ bf16 sBT[2][BN * BK];    // [n][k] transposed 2 x 8 KB

  const int tid  = threadIdx.x;
  const int lane = tid & 31;
  const int wave = tid >> 5;
  const int half = lane >> 4;
  const int l16  = lane & 15;
  const int wm   = wave >> 1;
  const int wn   = wave & 1;
  const int bm   = blockIdx.x * BM;
  const int bn   = blockIdx.y * BN;

  floatx8 acc[2][4] = {};
  uint4 breg[2];

  auto loadA_tile = [&](int buf, int k0) {
#if HAVE_TDM
    if (wave == 0)
      tdm_load_2d_bf16(&sA[buf][0], A + (size_t)bm * K + k0,
                       /*tile_d0=*/BK, /*tile_d1=*/BM, /*stride0=*/(uint64_t)K);
#else
#pragma unroll
    for (int i = 0; i < 2; ++i) {
      int u = tid + i * 256;
      int row = u >> 2, seg = u & 3;
      async_copy_b128(&sA[buf][row * BK + seg * 8],
                      A + (size_t)(bm + row) * K + k0 + seg * 8);
    }
#endif
  };
  auto loadB_global = [&](int k0) {
#pragma unroll
    for (int i = 0; i < 2; ++i) {
      int u = tid + i * 256;
      int krow = u >> 4, seg = u & 15;
      breg[i] = *(const uint4*)(Bm + (size_t)(k0 + krow) * N + bn + seg * 8);
    }
  };
  auto storeB_lds = [&](int buf) {
#pragma unroll
    for (int i = 0; i < 2; ++i) {
      int u = tid + i * 256;
      int krow = u >> 4, seg = u & 15;
      union { uint4 q; bf16 e[8]; } val;
      val.q = breg[i];
#pragma unroll
      for (int j = 0; j < 8; ++j)
        sBT[buf][(seg * 8 + j) * BK + krow] = val.e[j];
    }
  };

  loadA_tile(0, 0);
  loadB_global(0);
  storeB_lds(0);

  const int nk = K / BK;
  for (int kt = 0; kt < nk; ++kt) {
    const int cur = kt & 1;
    wait_tile_copies();     // issuing wave's DMA into sA[cur] complete
    __syncthreads();        // buffers [cur] globally ready; gates reuse of [cur^1]
    if (kt + 1 < nk) {
      loadA_tile(cur ^ 1, (kt + 1) * BK);   // DMA overlaps WMMA below
      loadB_global((kt + 1) * BK);
    }

    Frag af[2], bfr[4];
#pragma unroll
    for (int mi = 0; mi < 2; ++mi) {
      const bf16* arow = &sA[cur][(wm * 32 + mi * 16 + l16) * BK];
      af[mi].q[0] = *(const uint4*)(arow + half * 8);
      af[mi].q[1] = *(const uint4*)(arow + half * 8 + 16);
    }
#pragma unroll
    for (int ni = 0; ni < 4; ++ni) {
      const bf16* brow = &sBT[cur][(wn * 64 + ni * 16 + l16) * BK];
      bfr[ni].q[0] = *(const uint4*)(brow + half * 8);
      bfr[ni].q[1] = *(const uint4*)(brow + half * 8 + 16);
    }
#pragma unroll
    for (int mi = 0; mi < 2; ++mi)
#pragma unroll
      for (int ni = 0; ni < 4; ++ni)
        acc[mi][ni] = wmma_bf16(af[mi].v, bfr[ni].v, acc[mi][ni]);

    if (kt + 1 < nk) storeB_lds(cur ^ 1);   // B data arrived during WMMA
  }

#pragma unroll
  for (int mi = 0; mi < 2; ++mi)
#pragma unroll
    for (int ni = 0; ni < 4; ++ni)
#pragma unroll
      for (int r = 0; r < 8; ++r) {
        size_t row = (size_t)(bm + wm * 32 + mi * 16 + half * 8 + r);
        size_t col = (size_t)(bn + wn * 64 + ni * 16 + l16);
        if constexpr (OUT_F32) ((float*)C)[row * N + col] = acc[mi][ni][r];
        else                   ((bf16*)C)[row * N + col] = (bf16)acc[mi][ni][r];
      }
}

// ---------------------------------------------------------------------------
// RoPE (in place, bf16): rows = B*S, layout [row][head][128].
// ---------------------------------------------------------------------------
__global__ __launch_bounds__(256) void rope_kernel(
    bf16* __restrict__ qk, int rows, int S, int heads)
{
  int idx = blockIdx.x * blockDim.x + threadIdx.x;
  int total = rows * heads * 64;
  if (idx >= total) return;
  int i   = idx & 63;
  int h   = (idx >> 6) % heads;
  int row = idx / (64 * heads);
  int pos = row % S;
  float inv_freq = __powf(10000.0f, -(float)(2 * i) / 128.0f);
  float c, s;
  __sincosf((float)pos * inv_freq, &s, &c);
  bf16* base = qk + (size_t)row * heads * HEADDIM + (size_t)h * HEADDIM;
  float x1 = (float)base[i];
  float x2 = (float)base[i + 64];
  base[i]      = (bf16)(x1 * c - x2 * s);
  base[i + 64] = (bf16)(x2 * c + x1 * s);
}

// ---------------------------------------------------------------------------
// Flash attention (causal, GQA 16q/4kv heads, D=128).
// grid = (S/128, B*NHEADS), block = 256 (8 waves). Each wave: 16 q rows.
// KV streamed in 32-position tiles, double-buffered: K via TDM,
// V via pipelined VGPR transpose. Online softmax in registers.
// ---------------------------------------------------------------------------
__global__ __launch_bounds__(256) void flash_attn_kernel(
    const bf16* __restrict__ Q, const bf16* __restrict__ Km,
    const bf16* __restrict__ Vm, bf16* __restrict__ O, int B, int S)
{
  constexpr int KT = 32;
  __shared__ bf16 sK[2][KT * HEADDIM];     // [kpos][dim]   2 x 8 KB
  __shared__ bf16 sVT[2][HEADDIM * KT];    // [dim][kpos]   2 x 8 KB
  __shared__ bf16 sP[8][16 * KT];          // per-wave P tile, 8 KB

  const int tid  = threadIdx.x;
  const int lane = tid & 31;
  const int wave = tid >> 5;
  const int half = lane >> 4;
  const int l16  = lane & 15;

  const int bh  = blockIdx.y;
  const int b   = bh / NHEADS;
  const int h   = bh % NHEADS;
  const int kvh = h / (NHEADS / KVHEADS);
  const int q0  = blockIdx.x * 128 + wave * 16;

  const size_t qstride = (size_t)NHEADS * HEADDIM;
  const size_t kstride = (size_t)KVHEADS * HEADDIM;
  const bf16* qbase = Q  + (size_t)b * S * qstride + (size_t)h   * HEADDIM;
  const bf16* kbase = Km + (size_t)b * S * kstride + (size_t)kvh * HEADDIM;
  const bf16* vbase = Vm + (size_t)b * S * kstride + (size_t)kvh * HEADDIM;

  Frag qf[4];
#pragma unroll
  for (int c = 0; c < 4; ++c) {
    const bf16* qrow = qbase + (size_t)(q0 + l16) * qstride + c * 32;
    qf[c].q[0] = *(const uint4*)(qrow + half * 8);
    qf[c].q[1] = *(const uint4*)(qrow + half * 8 + 16);
  }

  floatx8 oacc[8] = {};
  float mrow[8], srow[8];
#pragma unroll
  for (int r = 0; r < 8; ++r) { mrow[r] = -INFINITY; srow[r] = 0.0f; }

  uint4 vreg[2];
  auto loadK_tile = [&](int buf, int kpos0) {
#if HAVE_TDM
    if (wave == 0)
      tdm_load_2d_bf16(&sK[buf][0], kbase + (size_t)kpos0 * kstride,
                       /*tile_d0=*/HEADDIM, /*tile_d1=*/KT, /*stride0=*/(uint64_t)kstride);
#else
#pragma unroll
    for (int i = 0; i < 2; ++i) {
      int u = tid + i * 256;
      int krow = u >> 4, seg = u & 15;
      async_copy_b128(&sK[buf][krow * HEADDIM + seg * 8],
                      kbase + (size_t)(kpos0 + krow) * kstride + seg * 8);
    }
#endif
  };
  auto loadV_global = [&](int kpos0) {
#pragma unroll
    for (int i = 0; i < 2; ++i) {
      int u = tid + i * 256;
      int vrow = u >> 4, seg = u & 15;
      vreg[i] = *(const uint4*)(vbase + (size_t)(kpos0 + vrow) * kstride + seg * 8);
    }
  };
  auto storeV_lds = [&](int buf) {
#pragma unroll
    for (int i = 0; i < 2; ++i) {
      int u = tid + i * 256;
      int vrow = u >> 4, seg = u & 15;
      union { uint4 q; bf16 e[8]; } val;
      val.q = vreg[i];
#pragma unroll
      for (int j = 0; j < 8; ++j)
        sVT[buf][(seg * 8 + j) * KT + vrow] = val.e[j];
    }
  };

  const float scale = 0.08838834764831845f;    // 1/sqrt(128)
  const int ktiles = (blockIdx.x + 1) * (128 / KT);

  loadK_tile(0, 0);
  loadV_global(0);
  storeV_lds(0);

  for (int kt = 0; kt < ktiles; ++kt) {
    const int cur = kt & 1;
    wait_tile_copies();
    __syncthreads();
    if (kt + 1 < ktiles) {
      loadK_tile(cur ^ 1, (kt + 1) * KT);
      loadV_global((kt + 1) * KT);
    }
    const int kpos0 = kt * KT;

    if (kpos0 <= q0 + 15) {
      // ---- scores S(16x32) = Q * K^T : 2 col-tiles x 4 dim-chunks = 8 WMMA
      floatx8 sc[2] = {};
#pragma unroll
      for (int c = 0; c < 4; ++c) {
#pragma unroll
        for (int t = 0; t < 2; ++t) {
          Frag kb;
          const bf16* krow_ = &sK[cur][(t * 16 + l16) * HEADDIM + c * 32];
          kb.q[0] = *(const uint4*)(krow_ + half * 8);
          kb.q[1] = *(const uint4*)(krow_ + half * 8 + 16);
          sc[t] = wmma_bf16(qf[c].v, kb.v, sc[t]);
        }
      }
      // ---- scale + causal mask
#pragma unroll
      for (int t = 0; t < 2; ++t) {
        int col = kpos0 + t * 16 + l16;
#pragma unroll
        for (int r = 0; r < 8; ++r) {
          float v = sc[t][r] * scale;
          if (col > q0 + half * 8 + r) v = -INFINITY;
          sc[t][r] = v;
        }
      }
      // ---- online softmax: row max (16-lane xor tree)
      float cand[8];
#pragma unroll
      for (int r = 0; r < 8; ++r) cand[r] = fmaxf(sc[0][r], sc[1][r]);
#pragma unroll
      for (int off = 8; off >= 1; off >>= 1)
#pragma unroll
        for (int r = 0; r < 8; ++r)
          cand[r] = fmaxf(cand[r], __shfl_xor(cand[r], off, 32));
      float alpha[8];
#pragma unroll
      for (int r = 0; r < 8; ++r) {
        float mn = fmaxf(mrow[r], cand[r]);
        alpha[r] = __expf(mrow[r] - mn);
        mrow[r] = mn;
      }
      // ---- P = exp(S - m) -> LDS (bf16), row sums
      float psum[8];
#pragma unroll
      for (int r = 0; r < 8; ++r) psum[r] = 0.0f;
      bf16* pw = &sP[wave][0];
#pragma unroll
      for (int t = 0; t < 2; ++t) {
        int colL = t * 16 + l16;
#pragma unroll
        for (int r = 0; r < 8; ++r) {
          float p = __expf(sc[t][r] - mrow[r]);
          psum[r] += p;
          pw[(half * 8 + r) * KT + colL] = (bf16)p;
        }
      }
#pragma unroll
      for (int off = 8; off >= 1; off >>= 1)
#pragma unroll
        for (int r = 0; r < 8; ++r)
          psum[r] += __shfl_xor(psum[r], off, 32);
#pragma unroll
      for (int r = 0; r < 8; ++r) srow[r] = srow[r] * alpha[r] + psum[r];
#pragma unroll
      for (int f = 0; f < 8; ++f)
#pragma unroll
        for (int r = 0; r < 8; ++r)
          oacc[f][r] *= alpha[r];
      // ---- O += P(16x32) * V(32x128): 8 WMMA (same-wave LDS is in-order)
      Frag pf;
      const bf16* prow = &sP[wave][l16 * KT];
      pf.q[0] = *(const uint4*)(prow + half * 8);
      pf.q[1] = *(const uint4*)(prow + half * 8 + 16);
#pragma unroll
      for (int f = 0; f < 8; ++f) {
        Frag vf;
        const bf16* vrow = &sVT[cur][(f * 16 + l16) * KT];
        vf.q[0] = *(const uint4*)(vrow + half * 8);
        vf.q[1] = *(const uint4*)(vrow + half * 8 + 16);
        oacc[f] = wmma_bf16(pf.v, vf.v, oacc[f]);
      }
    }

    if (kt + 1 < ktiles) storeV_lds(cur ^ 1);   // V data arrived during compute
  }

  float inv[8];
#pragma unroll
  for (int r = 0; r < 8; ++r) inv[r] = 1.0f / srow[r];
  bf16* obase = O + ((size_t)b * S + q0) * qstride + (size_t)h * HEADDIM;
#pragma unroll
  for (int f = 0; f < 8; ++f)
#pragma unroll
    for (int r = 0; r < 8; ++r)
      obase[(size_t)(half * 8 + r) * qstride + f * 16 + l16] =
          (bf16)(oacc[f][r] * inv[r]);
}

// ---------------------------------------------------------------------------
// Host launch
// ---------------------------------------------------------------------------
extern "C" void kernel_launch(void* const* d_in, const int* in_sizes, int n_in,
                              void* d_out, int out_size, void* d_ws, size_t ws_size,
                              hipStream_t stream)
{
  const int B = 2, S = 2048, H = 2048;
  const int M = B * S;                 // 4096
  const int KVW = KVHEADS * HEADDIM;   // 512

  const float* x  = (const float*)d_in[0];
  const float* wq = (const float*)d_in[1];
  const float* wk = (const float*)d_in[2];
  const float* wv = (const float*)d_in[3];
  const float* wo = (const float*)d_in[4];
  float* out = (float*)d_out;

  char* ws = (char*)d_ws;
  auto alloc = [&](size_t elems) {
    bf16* p = (bf16*)ws;
    ws += ((elems * sizeof(bf16) + 255) & ~(size_t)255);
    return p;
  };
  bf16* xb  = alloc((size_t)M * H);
  bf16* wqb = alloc((size_t)H * H);
  bf16* wkb = alloc((size_t)H * KVW);
  bf16* wvb = alloc((size_t)H * KVW);
  bf16* wob = alloc((size_t)H * H);
  bf16* Qb  = alloc((size_t)M * H);
  bf16* Kb  = alloc((size_t)M * KVW);
  bf16* Vb  = alloc((size_t)M * KVW);
  bf16* Ob  = xb;  // alias: xb dead after QKV GEMMs; rewritten next call

  auto cvt = [&](const float* src, bf16* dst, size_t n) {
    f32_to_bf16_kernel<<<(unsigned)((n + 255) / 256), 256, 0, stream>>>(src, dst, n);
  };
  cvt(x,  xb,  (size_t)M * H);
  cvt(wq, wqb, (size_t)H * H);
  cvt(wk, wkb, (size_t)H * KVW);
  cvt(wv, wvb, (size_t)H * KVW);
  cvt(wo, wob, (size_t)H * H);

  gemm_bf16_kernel<false><<<dim3(M / 128, H / 128),   256, 0, stream>>>(xb, wqb, Qb, M, H,   H);
  gemm_bf16_kernel<false><<<dim3(M / 128, KVW / 128), 256, 0, stream>>>(xb, wkb, Kb, M, KVW, H);
  gemm_bf16_kernel<false><<<dim3(M / 128, KVW / 128), 256, 0, stream>>>(xb, wvb, Vb, M, KVW, H);

  rope_kernel<<<(M * NHEADS  * 64) / 256, 256, 0, stream>>>(Qb, M, S, NHEADS);
  rope_kernel<<<(M * KVHEADS * 64) / 256, 256, 0, stream>>>(Kb, M, S, KVHEADS);

  flash_attn_kernel<<<dim3(S / 128, B * NHEADS), 256, 0, stream>>>(Qb, Kb, Vb, Ob, B, S);

  gemm_bf16_kernel<true><<<dim3(M / 128, H / 128), 256, 0, stream>>>(Ob, wob, out, M, H, H);
}
